// Attention_1262720385047
// MI455X (gfx1250) — matrix operand order repaired and checked by reading, not compiled, not run on hardware
//
#include <hip/hip_runtime.h>
#include <hip/hip_bf16.h>
#include <math.h>
#include <stdint.h>

// ---------------------------------------------------------------------------
// Bahdanau (MLP) attention for MI455X / gfx1250, wave32 + WMMA + async-LDS.
// Pipeline:
//   K1 (x2): qh = query@Wq^T + bq ; mh = memory@Wm^T        [v_wmma f16->f32]
//   K2     : attn[b,q,m] = sum_h v[h]*tanh(qh+mh), +mask    [async->LDS, v_tanh]
//   K3     : row softmax in place                           [VALU]
//   K4     : weighted_memory = weights @ memory             [v_wmma f16->f32]
// ---------------------------------------------------------------------------

typedef __attribute__((ext_vector_type(16))) _Float16 v16h;
typedef __attribute__((ext_vector_type(8)))  float    v8f;

#if defined(__has_builtin)
#  if __has_builtin(__builtin_amdgcn_tanhf)
#    define FAST_TANH(x) __builtin_amdgcn_tanhf(x)   // v_tanh_f32 (TRANS)
#  endif
#endif
#ifndef FAST_TANH
#  define FAST_TANH(x) tanhf(x)
#endif

#define B_    8
#define QLEN_ 256
#define MLEN_ 1024
#define QSZ_  256
#define MSZ_  256
#define HID_  128

// ---- WMMA fragment gathers (16x16x32 f16, wave32, contiguous k = 16*(lane/16) + elem) ----

// A (16x32, MxK), X row-major with leading dim ld; converts f32 -> f16.
// lanes 0-15: M=lane, VGPR0-7 hold K=0..15
// lanes16-31: M=lane-16, VGPR0-7 hold K=16..31
__device__ __forceinline__ v16h frag_a_rowmajor(const float* X, int ld, int row0, int k0) {
  const int lane = threadIdx.x & 31;
  const int g = lane >> 4;
  const int mr = lane & 15;
  const float* base = X + (size_t)(row0 + mr) * ld + k0;
  v16h a;
#pragma unroll
  for (int vg = 0; vg < 8; ++vg) {
    const int kb = g * 16 + vg * 2;
    float2 f = *(const float2*)(base + kb);
    a[2 * vg]     = (_Float16)f.x;
    a[2 * vg + 1] = (_Float16)f.y;
  }
  return a;
}

// B (32x16, KxN) where B[k][n] = W[n0+n][k0+k] (PyTorch-style (out,in) weight,
// K contiguous in memory per lane). lanes0-15: K=0..15, lanes16-31: K=16..31.
__device__ __forceinline__ v16h frag_b_wt(const float* W, int ld, int n0, int k0) {
  const int lane = threadIdx.x & 31;
  const int g = lane >> 4;
  const int n = lane & 15;
  const float* base = W + (size_t)(n0 + n) * ld + k0 + g * 16;
  v16h bf;
#pragma unroll
  for (int j = 0; j < 8; ++j) {
    float2 f = *(const float2*)(base + 2 * j);
    bf[2 * j]     = (_Float16)f.x;
    bf[2 * j + 1] = (_Float16)f.y;
  }
  return bf;
}

// B (32x16, KxN) where B[k][n] = M[k0+k][n0+n], M row-major (K strided by ld).
__device__ __forceinline__ v16h frag_b_rowmajor(const float* M, int ld, int k0, int n0) {
  const int lane = threadIdx.x & 31;
  const int g = lane >> 4;
  const int n = lane & 15;
  const float* base = M + (size_t)(k0 + g * 16) * ld + n0 + n;
  v16h bf;
#pragma unroll
  for (int e = 0; e < 16; ++e) bf[e] = (_Float16)base[(size_t)e * ld];
  return bf;
}

// ---- K1: C (R x 128) = X (R x 256) @ W(128 x 256)^T (+ bias) -----------------
__global__ void proj_gemm_wmma(const float* __restrict__ X, const float* __restrict__ W,
                               const float* __restrict__ bias, float* __restrict__ C, int R) {
  const int wave = threadIdx.x >> 5;              // 4 waves / block
  const int tile = blockIdx.x * 4 + wave;         // tiles = (R/16) * (128/16)
  const int rt = tile >> 3;
  const int ct = tile & 7;
  const int row0 = rt * 16, h0 = ct * 16;
  if (row0 >= R) return;                          // wave-uniform; EXEC stays full
  v8f acc = {};
#pragma unroll
  for (int k0 = 0; k0 < QSZ_; k0 += 32) {
    v16h a = frag_a_rowmajor(X, QSZ_, row0, k0);
    v16h b = frag_b_wt(W, QSZ_, h0, k0);
    acc = __builtin_amdgcn_wmma_f32_16x16x32_f16(false, a, false, b, (short)0, acc, false, false);
  }
  const int lane = threadIdx.x & 31;
  const int g = lane >> 4, n = lane & 15;
  const float bb = bias ? bias[h0 + n] : 0.0f;    // C VGPR r -> row M = r + 8*g
#pragma unroll
  for (int r = 0; r < 8; ++r)
    C[(size_t)(row0 + r + 8 * g) * HID_ + h0 + n] = acc[r] + bb;
}

// ---- K2: attn scores with tanh reduction over H, + mask ---------------------
// grid = B * 16 q-tiles * 4 m-segments = 512 blocks; 256 threads (8 waves).
// Thread layout: q = t>>4 (fixed per thread), m = (t&15) + 16*j, j=0..3.
// qh[q] and v are register-cached in 32-wide chunks -> 1 LDS float per tanh.
__global__ void attn_tanh_kernel(const float* __restrict__ qh, const float* __restrict__ mh,
                                 const float* __restrict__ v, const uint8_t* __restrict__ mask,
                                 float* __restrict__ wts) {
  __shared__ alignas(16) float mh_s[64][132];   // stride 132 floats: 16B rows, no conflicts
  __shared__ alignas(16) float qh_s[16][132];
  __shared__ alignas(16) float v_s[128];
  const int bid = blockIdx.x;
  const int b   = bid >> 6;
  const int qt  = (bid >> 2) & 15;
  const int seg = bid & 3;                      // covers m in [seg*256, seg*256+256)
  const int q0  = qt * 16;
  const int t   = threadIdx.x;
  const int q   = t >> 4;                       // 0..15 (16 lanes share a q -> broadcast)
  const int ml  = t & 15;

  // stage qh tile (16x128) and v into LDS (float4, coalesced, sync path)
#pragma unroll
  for (int j = 0; j < 2; ++j) {
    const int idx = t + j * 256;                // 512 float4 = 16*128 floats
    const int qr = idx >> 5;
    const int h4 = (idx & 31) * 4;
    *(float4*)&qh_s[qr][h4] =
        *(const float4*)&qh[((size_t)(b * QLEN_ + q0 + qr)) * HID_ + h4];
  }
  if (t < 32) *(float4*)&v_s[t * 4] = *(const float4*)&v[t * 4];

  // LDS byte offset of mh_s (flat LDS aperture: low 32 bits = LDS address)
  const unsigned lds_mh = (unsigned)(uintptr_t)&mh_s[0][0];

  for (int mt = 0; mt < 4; ++mt) {
    const int m0 = seg * 256 + mt * 64;
    __syncthreads();                            // previous tile fully consumed

    // async memory -> LDS staging of the 64x128 mh tile (no VGPR round trip)
#pragma unroll
    for (int j = 0; j < 8; ++j) {
      const int idx = t + j * 256;              // 2048 float4
      const int m  = idx >> 5;
      const int h4 = (idx & 31) * 4;
      const unsigned lo = lds_mh + (unsigned)((m * 132 + h4) * 4);
      const float* g = &mh[((size_t)(b * MLEN_ + m0 + m)) * HID_ + h4];
      asm volatile("global_load_async_to_lds_b128 %0, %1, off"
                   :: "v"(lo), "v"(g) : "memory");
    }
    asm volatile("s_wait_asynccnt 0x0" ::: "memory");  // own wave's async done
    __syncthreads();                                   // all waves' async done

    float acc[4] = {0.f, 0.f, 0.f, 0.f};
    for (int hc = 0; hc < 128; hc += 32) {
      float4 qv[8], vv[8];
#pragma unroll
      for (int i = 0; i < 8; ++i) {             // broadcast LDS reads -> registers
        qv[i] = *(const float4*)&qh_s[q][hc + 4 * i];
        vv[i] = *(const float4*)&v_s[hc + 4 * i];
      }
#pragma unroll
      for (int j = 0; j < 4; ++j) {
        const int m = ml + 16 * j;
#pragma unroll
        for (int i = 0; i < 8; ++i) {
          const float4 mv = *(const float4*)&mh_s[m][hc + 4 * i];
          acc[j] += vv[i].x * FAST_TANH(qv[i].x + mv.x);  // v_tanh TRANS co-issues
          acc[j] += vv[i].y * FAST_TANH(qv[i].y + mv.y);
          acc[j] += vv[i].z * FAST_TANH(qv[i].z + mv.z);
          acc[j] += vv[i].w * FAST_TANH(qv[i].w + mv.w);
        }
      }
    }
#pragma unroll
    for (int j = 0; j < 4; ++j) {
      const int mg = m0 + ml + 16 * j;
      const float val = mask[(size_t)b * MLEN_ + mg] ? -1.0e30f : acc[j];
      wts[((size_t)(b * QLEN_ + q0 + q)) * MLEN_ + mg] = val;
    }
  }
}

// ---- K3: in-place softmax over rows of 1024 ---------------------------------
__global__ void softmax_kernel(float* __restrict__ wts) {
  __shared__ float red[256];
  float* w = wts + (size_t)blockIdx.x * MLEN_;
  const int t = threadIdx.x;
  float vals[4];
  float mx = -3.4e38f;
#pragma unroll
  for (int j = 0; j < 4; ++j) { vals[j] = w[t + j * 256]; mx = fmaxf(mx, vals[j]); }
  red[t] = mx; __syncthreads();
  for (int s = 128; s > 0; s >>= 1) {
    if (t < s) red[t] = fmaxf(red[t], red[t + s]);
    __syncthreads();
  }
  mx = red[0]; __syncthreads();
  float sum = 0.0f;
#pragma unroll
  for (int j = 0; j < 4; ++j) { vals[j] = __expf(vals[j] - mx); sum += vals[j]; }
  red[t] = sum; __syncthreads();
  for (int s = 128; s > 0; s >>= 1) {
    if (t < s) red[t] += red[t + s];
    __syncthreads();
  }
  const float inv = 1.0f / red[0];
#pragma unroll
  for (int j = 0; j < 4; ++j) w[t + j * 256] = vals[j] * inv;
}

// ---- K4: weighted_memory[b] (256x256) = weights[b](256x1024) @ memory[b](1024x256)
__global__ void out_gemm_wmma(const float* __restrict__ wts, const float* __restrict__ mem,
                              float* __restrict__ out) {
  const int wave = threadIdx.x >> 5;
  const int tile = blockIdx.x * 4 + wave;       // 8 * 16 * 16 = 2048 tiles
  const int b  = tile >> 8;
  const int qt = (tile >> 4) & 15;
  const int dt = tile & 15;
  const float* A  = wts + (size_t)b * QLEN_ * MLEN_;
  const float* Bm = mem + (size_t)b * MLEN_ * MSZ_;
  float* C = out + (size_t)b * QLEN_ * MSZ_;
  v8f acc = {};
  for (int k0 = 0; k0 < MLEN_; k0 += 32) {      // 32 K per wmma, 32 steps
    v16h a  = frag_a_rowmajor(A, MLEN_, qt * 16, k0);
    v16h bf = frag_b_rowmajor(Bm, MSZ_, k0, dt * 16);
    acc = __builtin_amdgcn_wmma_f32_16x16x32_f16(false, a, false, bf, (short)0, acc, false, false);
  }
  const int lane = threadIdx.x & 31;
  const int g = lane >> 4, n = lane & 15;
#pragma unroll
  for (int r = 0; r < 8; ++r)
    C[(size_t)(qt * 16 + r + 8 * g) * MSZ_ + dt * 16 + n] = acc[r];
}

// ---------------------------------------------------------------------------
extern "C" void kernel_launch(void* const* d_in, const int* in_sizes, int n_in,
                              void* d_out, int out_size, void* d_ws, size_t ws_size,
                              hipStream_t stream) {
  (void)in_sizes; (void)n_in; (void)out_size; (void)ws_size;
  const float*   query  = (const float*)d_in[0];   // (8,256,256)
  const float*   memory = (const float*)d_in[1];   // (8,1024,256)
  const float*   Wq     = (const float*)d_in[2];   // (128,256)
  const float*   bq     = (const float*)d_in[3];   // (128)
  const float*   Wm     = (const float*)d_in[4];   // (128,256)
  const float*   v      = (const float*)d_in[5];   // (128)
  const uint8_t* mask   = (const uint8_t*)d_in[6]; // (8,1024) bool (1 byte/elem)

  float* wm_out  = (float*)d_out;                              // (8,256,256)
  float* wts_out = (float*)d_out + (size_t)B_ * QLEN_ * MSZ_;  // (8,256,1024)

  float* qh = (float*)d_ws;                        // (2048,128) = 1 MB
  float* mh = qh + (size_t)(B_ * QLEN_) * HID_;    // (8192,128) = 4 MB

  // K1: projections (WMMA). 4 wave-tiles per 128-thread block.
  proj_gemm_wmma<<<256,  128, 0, stream>>>(query,  Wq, bq,      qh, B_ * QLEN_);
  proj_gemm_wmma<<<1024, 128, 0, stream>>>(memory, Wm, nullptr, mh, B_ * MLEN_);
  // K2: tanh-score stage (TRANS-bound), masked scores -> weights buffer.
  attn_tanh_kernel<<<512, 256, 0, stream>>>(qh, mh, v, mask, wts_out);
  // K3: softmax rows in place.
  softmax_kernel<<<B_ * QLEN_, 256, 0, stream>>>(wts_out);
  // K4: output GEMM (WMMA).
  out_gemm_wmma<<<512, 128, 0, stream>>>(wts_out, memory, wm_out);
}
